// SliceAttentionIrregularMesh_41910290874676
// MI455X (gfx1250) — matrix operand order repaired and checked by reading, not compiled, not run on hardware
//
#include <hip/hip_runtime.h>
#include <hip/hip_bf16.h>

#define NTOK   200000
#define HDIM   256
#define NHEADS 8
#define DH     32
#define GS     64
#define BM     128

typedef __attribute__((ext_vector_type(8)))  unsigned int v8u;
typedef __attribute__((ext_vector_type(8)))  float        v8f;
typedef __attribute__((ext_vector_type(16))) __bf16       v16bf;

union FragU { v8u u; v16bf b; uint4 q[2]; };

__device__ __forceinline__ unsigned short f2bf(float f) {
  unsigned int u = __float_as_uint(f);
  u += 0x7FFFu + ((u >> 16) & 1u);          // round-to-nearest-even
  return (unsigned short)(u >> 16);
}

__device__ __forceinline__ v8f vzero8() {
  v8f z = {0.f,0.f,0.f,0.f,0.f,0.f,0.f,0.f};
  return z;
}

// A fragment: 16x32 bf16 (MxK), row-major source with leading dim ldm (elements).
// Lane L (0-15): M=L, dwords 0-3 hold K=0..7, dwords 4-7 hold K=16..23.
// Lane L (16-31): M=L-16, K offset +8 (K=8..15 / 24..31).
__device__ __forceinline__ v16bf frag_a(const unsigned short* base, int ldm) {
  int lane = threadIdx.x & 31;
  int m  = lane & 15;
  int kb = (lane >> 4) << 3;                 // 0 or 8
  const unsigned short* p = base + m * ldm + kb;
  FragU f;
  f.q[0] = *(const uint4*)(p);
  f.q[1] = *(const uint4*)(p + 16);
  return f.b;
}

// B fragment: 32x16 bf16 (KxN), where B[k][nj] = W[(ncol0+nj)*ldw + k0 + k]
// Lane L (0-15): N=L, dwords 0-3 hold K=0..7, dwords 4-7 hold K=16..23; lanes 16-31: N=L-16, K=8..15 / 24..31.
__device__ __forceinline__ v16bf frag_b(const unsigned short* W, int ldw,
                                        int ncol0, int k0) {
  int lane = threadIdx.x & 31;
  int nj = lane & 15;
  int kb = (lane >> 4) << 3;                 // 0 or 8
  const unsigned short* p = W + (ncol0 + nj) * ldw + k0 + kb;
  FragU f;
  f.q[0] = *(const uint4*)(p);
  f.q[1] = *(const uint4*)(p + 16);
  return f.b;
}

__device__ __forceinline__ v8f wmma_bf16(v16bf a, v16bf b, v8f c) {
  return __builtin_amdgcn_wmma_f32_16x16x32_bf16(false, a, false, b,
                                                 (short)0, c, false, false);
}

// ---------------------------------------------------------------- prep ------
__global__ __launch_bounds__(256)
void prep_kernel(const float* __restrict__ Wx, const float* __restrict__ Wfx,
                 const float* __restrict__ Wsl, const float* __restrict__ Wout,
                 unsigned short* __restrict__ WxB, unsigned short* __restrict__ WfxB,
                 unsigned short* __restrict__ WslB, unsigned short* __restrict__ WoutB) {
  int i = blockIdx.x * blockDim.x + threadIdx.x;
  if (i < HDIM * HDIM) {
    WxB[i]   = f2bf(Wx[i]);
    WfxB[i]  = f2bf(Wfx[i]);
    WoutB[i] = f2bf(Wout[i]);
  }
  if (i < GS * DH) WslB[i] = f2bf(Wsl[i]);
}

// ---------------------------------------------------------------- pass 1 ----
// Per 128-row tile: x@WxT+bx, x@WfxT+bfx, slice softmax, store sw (bf16),
// partial slice_token (WMMA, K=128) and partial slice_norm per block.
__global__ __launch_bounds__(256)
void pass1_kernel(const float* __restrict__ x,
                  const float* __restrict__ bx,
                  const float* __restrict__ bfx,
                  const float* __restrict__ bslice,
                  const float* __restrict__ temperature,
                  const unsigned short* __restrict__ WxB,
                  const unsigned short* __restrict__ WfxB,
                  const unsigned short* __restrict__ WslB,
                  unsigned short* __restrict__ sw,
                  float* __restrict__ tokenPartial,
                  float* __restrict__ normPartial) {
  __shared__ __align__(16) unsigned short xlds[BM][HDIM];       // 64KB
  __shared__ __align__(16) unsigned short ylds[8][16][64];      // 16KB (x_mid chunk per wave)
  __shared__ __align__(16) unsigned short fxlds[64][BM];        // 16KB (fx chunk, transposed)
  __shared__ __align__(16) unsigned short wlds[2][GS][BM];      // 32KB (w transposed)
  __shared__ float normw[8][NHEADS * GS];                       // 16KB

  const int tid = threadIdx.x;
  const int wave = tid >> 5;
  const int lane = tid & 31;
  const int laneHalf = lane >> 4;
  const int lcol = lane & 15;
  const int row0 = blockIdx.x * BM;
  const int wrow0 = wave * 16;

  // load + convert x tile to bf16 LDS (half row per thread)
  {
    int r = tid >> 1, half = tid & 1;
    int gr = row0 + r; if (gr >= NTOK) gr = NTOK - 1;
    const float4* src = (const float4*)(x + (size_t)gr * HDIM + half * 128);
    unsigned int* dst = (unsigned int*)&xlds[r][half * 128];
    #pragma unroll
    for (int j = 0; j < 32; ++j) {
      float4 v = src[j];
      dst[j * 2]     = ((unsigned int)f2bf(v.y) << 16) | f2bf(v.x);
      dst[j * 2 + 1] = ((unsigned int)f2bf(v.w) << 16) | f2bf(v.z);
    }
  }
  __syncthreads();

  #pragma unroll 1
  for (int chunk = 0; chunk < 4; ++chunk) {
    const int ncol0 = chunk * 64;
    // ---- main dual GEMM: Y1 = x@WxT, Y2 = x@WfxT (16 rows x 64 cols) ----
    v8f acc1[4], acc2[4];
    #pragma unroll
    for (int t = 0; t < 4; ++t) { acc1[t] = vzero8(); acc2[t] = vzero8(); }
    #pragma unroll
    for (int ks = 0; ks < 8; ++ks) {
      int k0 = ks * 32;
      v16bf A = frag_a(&xlds[wrow0][k0], HDIM);
      #pragma unroll
      for (int t = 0; t < 4; ++t) {
        acc1[t] = wmma_bf16(A, frag_b(WxB,  HDIM, ncol0 + t * 16, k0), acc1[t]);
        acc2[t] = wmma_bf16(A, frag_b(WfxB, HDIM, ncol0 + t * 16, k0), acc2[t]);
      }
    }
    // ---- bias + stage x_mid (row-major) and fx (transposed) ----
    #pragma unroll
    for (int t = 0; t < 4; ++t) {
      int c = ncol0 + t * 16 + lcol;
      float b1 = bx[c], b2 = bfx[c];
      int ccl = t * 16 + lcol;
      #pragma unroll
      for (int r = 0; r < 8; ++r) {
        int m = r + laneHalf * 8;
        ylds[wave][m][ccl]    = f2bf(acc1[t][r] + b1);
        fxlds[ccl][wrow0 + m] = f2bf(acc2[t][r] + b2);
      }
    }
    __syncthreads();

    // ---- logits + softmax for the chunk's two heads ----
    #pragma unroll 1
    for (int hh = 0; hh < 2; ++hh) {
      int head = chunk * 2 + hh;
      v16bf Al = frag_a(&ylds[wave][0][hh * 32], 64);
      v8f lac[4];
      #pragma unroll
      for (int gt = 0; gt < 4; ++gt)
        lac[gt] = wmma_bf16(Al, frag_b(WslB, DH, gt * 16, 0), vzero8());

      float tc = temperature[head];
      tc = fminf(fmaxf(tc, 0.1f), 5.0f);
      float invt = 1.0f / tc;
      #pragma unroll
      for (int gt = 0; gt < 4; ++gt) {
        float bs = bslice[gt * 16 + lcol];
        #pragma unroll
        for (int r = 0; r < 8; ++r) lac[gt][r] = (lac[gt][r] + bs) * invt;
      }
      float rmax[8], rinv[8];
      #pragma unroll
      for (int r = 0; r < 8; ++r) {
        float mv = fmaxf(fmaxf(lac[0][r], lac[1][r]), fmaxf(lac[2][r], lac[3][r]));
        #pragma unroll
        for (int s = 1; s <= 8; s <<= 1) mv = fmaxf(mv, __shfl_xor(mv, s, 32));
        rmax[r] = mv;
      }
      #pragma unroll
      for (int gt = 0; gt < 4; ++gt)
        #pragma unroll
        for (int r = 0; r < 8; ++r) lac[gt][r] = __expf(lac[gt][r] - rmax[r]);
      #pragma unroll
      for (int r = 0; r < 8; ++r) {
        float s = lac[0][r] + lac[1][r] + lac[2][r] + lac[3][r];
        #pragma unroll
        for (int t2 = 1; t2 <= 8; t2 <<= 1) s += __shfl_xor(s, t2, 32);
        rinv[r] = 1.0f / s;
      }
      #pragma unroll
      for (int gt = 0; gt < 4; ++gt) {
        int g = gt * 16 + lcol;
        float npart = 0.f;
        #pragma unroll
        for (int r = 0; r < 8; ++r) {
          int m = r + laneHalf * 8;
          int n = row0 + wrow0 + m;
          float w = lac[gt][r] * rinv[r];
          bool valid = (n < NTOK);
          if (valid) sw[(size_t)n * (NHEADS * GS) + head * GS + g] = f2bf(w);
          wlds[hh][g][wrow0 + m] = valid ? f2bf(w) : (unsigned short)0;
          npart += valid ? w : 0.f;
        }
        npart += __shfl_xor(npart, 16, 32);   // add rows 8..15 half
        if (lane < 16) normw[wave][head * GS + g] = npart;
      }
    }
    __syncthreads();

    // ---- token GEMM: token[h][g][c] += sum_rows w[row][g]*fx[row][c], K=128 ----
    {
      int hic  = wave >> 2;
      int head = chunk * 2 + hic;
      int gt   = wave & 3;
      v8f tac[2]; tac[0] = vzero8(); tac[1] = vzero8();
      #pragma unroll
      for (int ks = 0; ks < 4; ++ks) {
        v16bf A = frag_a(&wlds[hic][gt * 16][ks * 32], BM);
        #pragma unroll
        for (int nt = 0; nt < 2; ++nt)
          tac[nt] = wmma_bf16(A, frag_b(&fxlds[0][0], BM, hic * 32 + nt * 16, ks * 32), tac[nt]);
      }
      float* tp = tokenPartial + (size_t)blockIdx.x * (NHEADS * GS * DH);
      #pragma unroll
      for (int nt = 0; nt < 2; ++nt)
        #pragma unroll
        for (int r = 0; r < 8; ++r) {
          int grow = gt * 16 + r + laneHalf * 8;
          int c    = nt * 16 + lcol;
          tp[(head * GS + grow) * DH + c] = tac[nt][r];
        }
    }
    __syncthreads();
  }

  // fixed-order norm partial reduction (deterministic)
  float s0 = 0.f, s1 = 0.f;
  #pragma unroll
  for (int w = 0; w < 8; ++w) { s0 += normw[w][tid]; s1 += normw[w][tid + 256]; }
  normPartial[(size_t)blockIdx.x * (NHEADS * GS) + tid]       = s0;
  normPartial[(size_t)blockIdx.x * (NHEADS * GS) + tid + 256] = s1;
}

// ---------------------------------------------------------------- reduce ----
__global__ __launch_bounds__(256)
void reduce_kernel(const float* __restrict__ tokenPartial,
                   const float* __restrict__ normPartial,
                   int nblocks,
                   float* __restrict__ token,
                   float* __restrict__ norm) {
  int idx = blockIdx.x * blockDim.x + threadIdx.x;
  if (idx < NHEADS * GS * DH) {
    float s = 0.f;
    for (int b = 0; b < nblocks; ++b)
      s += tokenPartial[(size_t)b * (NHEADS * GS * DH) + idx];
    token[idx] = s;
  } else if (idx < NHEADS * GS * DH + NHEADS * GS) {
    int j = idx - NHEADS * GS * DH;
    float s = 0.f;
    for (int b = 0; b < nblocks; ++b)
      s += normPartial[(size_t)b * (NHEADS * GS) + j];
    norm[j] = s;
  }
}

// ---------------------------------------------------------------- pass 2 ----
// Tiny 8-head attention over 64 slice tokens. One workgroup.
__global__ __launch_bounds__(256)
void pass2_kernel(const float* __restrict__ token,
                  const float* __restrict__ norm,
                  const float* __restrict__ Wq,
                  const float* __restrict__ Wk,
                  const float* __restrict__ Wv,
                  unsigned short* __restrict__ osT) {
  __shared__ float stl[NHEADS * GS * DH];
  __shared__ float kl [NHEADS * GS * DH];
  __shared__ float vl [NHEADS * GS * DH];
  const int tid = threadIdx.x;
  for (int i = tid; i < NHEADS * GS * DH; i += 256) {
    int hg = i >> 5;
    stl[i] = token[i] / (norm[hg] + 1e-5f);
  }
  __syncthreads();
  for (int p = 0; p < 2; ++p) {
    int hg = tid + p * 256;
    const float* st = &stl[hg * DH];
    for (int e = 0; e < DH; ++e) {
      float sk = 0.f, sv = 0.f;
      for (int d = 0; d < DH; ++d) {
        float s = st[d];
        sk += s * Wk[e * DH + d];
        sv += s * Wv[e * DH + d];
      }
      kl[hg * DH + e] = sk;
      vl[hg * DH + e] = sv;
    }
  }
  __syncthreads();
  const float scale = 0.17677669529663687f;   // 32^-0.5
  for (int p = 0; p < 2; ++p) {
    int hg = tid + p * 256;
    int h = hg >> 6, g = hg & 63;
    const float* st = &stl[hg * DH];
    float q[DH];
    for (int e = 0; e < DH; ++e) {
      float s = 0.f;
      for (int d = 0; d < DH; ++d) s += st[d] * Wq[e * DH + d];
      q[e] = s;
    }
    float dots[GS];
    float mx = -3.0e38f;
    for (int j = 0; j < GS; ++j) {
      const float* kj = &kl[(h * GS + j) * DH];
      float s = 0.f;
      for (int d = 0; d < DH; ++d) s += q[d] * kj[d];
      s *= scale;
      dots[j] = s;
      mx = fmaxf(mx, s);
    }
    float ssum = 0.f;
    for (int j = 0; j < GS; ++j) { dots[j] = __expf(dots[j] - mx); ssum += dots[j]; }
    float inv = 1.0f / ssum;
    for (int c = 0; c < DH; ++c) {
      float o = 0.f;
      for (int j = 0; j < GS; ++j) o += dots[j] * vl[(h * GS + j) * DH + c];
      osT[(h * DH + c) * GS + g] = f2bf(o * inv);   // stored transposed [h][c][g]
    }
  }
}

// ---------------------------------------------------------------- pass 3 ----
// out = (sum_g sw[n,h,g]*out_slice[h,g,:]) @ WoutT + bout
__global__ __launch_bounds__(256)
void pass3_kernel(const unsigned short* __restrict__ sw,
                  const unsigned short* __restrict__ osT,
                  const unsigned short* __restrict__ WoutB,
                  const float* __restrict__ bout,
                  float* __restrict__ out) {
  __shared__ __align__(16) unsigned short swl[BM][NHEADS * GS]; // 128KB
  __shared__ __align__(16) unsigned short oxl[BM][HDIM];        // 64KB
  const int tid = threadIdx.x;
  const int wave = tid >> 5, lane = tid & 31;
  const int laneHalf = lane >> 4, lcol = lane & 15;
  const int row0 = blockIdx.x * BM, wrow0 = wave * 16;

  { // load sw tile (coalesced b128)
    int r = tid >> 1, half = tid & 1;
    int gr = row0 + r; if (gr >= NTOK) gr = NTOK - 1;
    const uint4* src = (const uint4*)(sw + (size_t)gr * (NHEADS * GS) + half * 256);
    uint4* dst = (uint4*)&swl[r][half * 256];
    #pragma unroll
    for (int j = 0; j < 32; ++j) dst[j] = src[j];
  }
  __syncthreads();

  // combine: out_x[n, h*32+c] = sum_g sw[n,h,g] * out_slice[h,g,c]
  #pragma unroll 1
  for (int h = 0; h < NHEADS; ++h) {
    v8f cac[2]; cac[0] = vzero8(); cac[1] = vzero8();
    #pragma unroll
    for (int ks = 0; ks < 2; ++ks) {
      v16bf A = frag_a(&swl[wrow0][h * GS + ks * 32], NHEADS * GS);
      #pragma unroll
      for (int nt = 0; nt < 2; ++nt)
        cac[nt] = wmma_bf16(A, frag_b(osT + h * DH * GS, GS, nt * 16, ks * 32), cac[nt]);
    }
    #pragma unroll
    for (int nt = 0; nt < 2; ++nt)
      #pragma unroll
      for (int r = 0; r < 8; ++r) {
        int m = r + laneHalf * 8;
        oxl[wrow0 + m][h * 32 + nt * 16 + lcol] = f2bf(cac[nt][r]);
      }
  }
  __syncthreads();

  // final GEMM vs WoutT + bias
  #pragma unroll 1
  for (int chunk = 0; chunk < 4; ++chunk) {
    v8f oac[4];
    #pragma unroll
    for (int t = 0; t < 4; ++t) oac[t] = vzero8();
    #pragma unroll
    for (int ks = 0; ks < 8; ++ks) {
      v16bf A = frag_a(&oxl[wrow0][ks * 32], HDIM);
      #pragma unroll
      for (int nt = 0; nt < 4; ++nt)
        oac[nt] = wmma_bf16(A, frag_b(WoutB, HDIM, chunk * 64 + nt * 16, ks * 32), oac[nt]);
    }
    #pragma unroll
    for (int nt = 0; nt < 4; ++nt) {
      int col = chunk * 64 + nt * 16 + lcol;
      float bo = bout[col];
      #pragma unroll
      for (int r = 0; r < 8; ++r) {
        int n = row0 + wrow0 + r + laneHalf * 8;
        if (n < NTOK) out[(size_t)n * HDIM + col] = oac[nt][r] + bo;
      }
    }
  }
}

// ---------------------------------------------------------------- launch ----
extern "C" void kernel_launch(void* const* d_in, const int* in_sizes, int n_in,
                              void* d_out, int out_size, void* d_ws, size_t ws_size,
                              hipStream_t stream) {
  (void)in_sizes; (void)n_in; (void)out_size; (void)ws_size;
  const float* x    = (const float*)d_in[0];
  const float* Wx   = (const float*)d_in[1];
  const float* bx   = (const float*)d_in[2];
  const float* Wfx  = (const float*)d_in[3];
  const float* bfx  = (const float*)d_in[4];
  const float* Wsl  = (const float*)d_in[5];
  const float* bsl  = (const float*)d_in[6];
  const float* Wq   = (const float*)d_in[7];
  const float* Wk   = (const float*)d_in[8];
  const float* Wv   = (const float*)d_in[9];
  const float* Wout = (const float*)d_in[10];
  const float* bout = (const float*)d_in[11];
  const float* temp = (const float*)d_in[12];
  float* out = (float*)d_out;

  const int nblocks = (NTOK + BM - 1) / BM;   // 1563

  char* ws = (char*)d_ws;
  size_t off = 0;
  unsigned short* WxB   = (unsigned short*)(ws + off); off += 131072;
  unsigned short* WfxB  = (unsigned short*)(ws + off); off += 131072;
  unsigned short* WslB  = (unsigned short*)(ws + off); off += 4096;
  unsigned short* WoutB = (unsigned short*)(ws + off); off += 131072;
  float* token          = (float*)(ws + off);          off += 65536;
  float* norm           = (float*)(ws + off);          off += 2048;
  unsigned short* osT   = (unsigned short*)(ws + off); off += 32768;
  float* tokenPartial   = (float*)(ws + off);          off += (size_t)nblocks * NHEADS * GS * DH * 4;
  float* normPartial    = (float*)(ws + off);          off += (size_t)nblocks * NHEADS * GS * 4;
  unsigned short* sw    = (unsigned short*)(ws + off); // N * 512 bf16

  prep_kernel<<<(HDIM * HDIM + 255) / 256, 256, 0, stream>>>(
      Wx, Wfx, Wsl, Wout, WxB, WfxB, WslB, WoutB);
  pass1_kernel<<<nblocks, 256, 0, stream>>>(
      x, bx, bfx, bsl, temp, WxB, WfxB, WslB, sw, tokenPartial, normPartial);
  reduce_kernel<<<(NHEADS * GS * DH + NHEADS * GS + 255) / 256, 256, 0, stream>>>(
      tokenPartial, normPartial, nblocks, token, norm);
  pass2_kernel<<<1, 256, 0, stream>>>(token, norm, Wq, Wk, Wv, osT);
  pass3_kernel<<<nblocks, 256, 0, stream>>>(sw, osT, WoutB, bout, out);
}
